// Self_Attn_Layer_49039936586065
// MI455X (gfx1250) — matrix operand order repaired and checked by reading, not compiled, not run on hardware
//
#include <hip/hip_runtime.h>
#include <hip/hip_bf16.h>
#include <math.h>

typedef __attribute__((ext_vector_type(16))) __bf16 v16bf;
typedef __attribute__((ext_vector_type(8)))  __bf16 v8bf;
typedef __attribute__((ext_vector_type(4)))  __bf16 v4bf;
typedef __attribute__((ext_vector_type(8)))  float  v8f;
typedef __attribute__((ext_vector_type(4)))  float  v4f;

#define BATCH 4
#define CCH   256
#define NPIX  4096   // 64*64
#define DKQ   32     // C/8

// ---------------------------------------------------------------------------
// Kernel 1: q/k projection. q[b,n,d] = sum_c Wq[d,c]*x[b,c,n] + bq[d]
// Output rows of 32 bf16 match WMMA A/B fragment addressing exactly.
// ---------------------------------------------------------------------------
__global__ __launch_bounds__(256)
void proj_qk_kernel(const float* __restrict__ x,
                    const float* __restrict__ Wq, const float* __restrict__ bq,
                    const float* __restrict__ Wk, const float* __restrict__ bk,
                    __bf16* __restrict__ qbf, __bf16* __restrict__ kbf) {
    __shared__ float lds[CCH * DKQ];          // 32 KB, [c][d] (transposed W)
    const int t  = threadIdx.x;
    const int b  = blockIdx.x / (NPIX / 256);
    const int n  = (blockIdx.x % (NPIX / 256)) * 256 + t;
    const float* xb = x + (size_t)b * CCH * NPIX;

    float acc[DKQ];

    // ---- Q ----
    for (int d = 0; d < DKQ; ++d) lds[t * DKQ + d] = Wq[d * CCH + t];
    __syncthreads();
    for (int d = 0; d < DKQ; ++d) acc[d] = bq[d];
    for (int c = 0; c < CCH; ++c) {
        float xv = xb[(size_t)c * NPIX + n];          // coalesced across t
        const float* wrow = &lds[c * DKQ];            // LDS broadcast
        #pragma unroll
        for (int d = 0; d < DKQ; ++d) acc[d] += wrow[d] * xv;
    }
    {
        __bf16* qrow = qbf + ((size_t)b * NPIX + n) * DKQ;
        #pragma unroll
        for (int d = 0; d < DKQ; ++d) qrow[d] = (__bf16)acc[d];
    }
    __syncthreads();

    // ---- K ----
    for (int d = 0; d < DKQ; ++d) lds[t * DKQ + d] = Wk[d * CCH + t];
    __syncthreads();
    for (int d = 0; d < DKQ; ++d) acc[d] = bk[d];
    for (int c = 0; c < CCH; ++c) {
        float xv = xb[(size_t)c * NPIX + n];
        const float* wrow = &lds[c * DKQ];
        #pragma unroll
        for (int d = 0; d < DKQ; ++d) acc[d] += wrow[d] * xv;
    }
    {
        __bf16* krow = kbf + ((size_t)b * NPIX + n) * DKQ;
        #pragma unroll
        for (int d = 0; d < DKQ; ++d) krow[d] = (__bf16)acc[d];
    }
}

// ---------------------------------------------------------------------------
// Kernel 2: v projection (bf16 out). v[b,d,n] = sum_c Wv[d,c]*x[b,c,n] + bv[d]
// x tile is streamed global->LDS with async-tensor copies, double buffered:
// chunk (16 channels x 256 pixels = 16KB) i+1 is in flight while chunk i is
// consumed.  global_load_async_to_lds_b128 + s_wait_asynccnt (gfx1250).
// ---------------------------------------------------------------------------
#define VCHUNK 16                      // channels per pipeline stage
#define NSTAGE (CCH / VCHUNK)          // 16 stages

__global__ __launch_bounds__(256)
void proj_v_kernel(const float* __restrict__ x,
                   const float* __restrict__ Wv, const float* __restrict__ bv,
                   __bf16* __restrict__ vbf) {
    __shared__ float wv[CCH * 32];                 // 32 KB Wv tile, [c][dd]
    __shared__ float xs[2][VCHUNK * 256];          // 2 x 16 KB x chunks
    const int t  = threadIdx.x;
    const int b  = blockIdx.x / (NPIX / 256);
    const int n0 = (blockIdx.x % (NPIX / 256)) * 256;
    const int n  = n0 + t;
    const int d0 = blockIdx.y * 32;
    const float* xb = x + (size_t)b * CCH * NPIX;

    for (int dd = 0; dd < 32; ++dd) wv[t * 32 + dd] = Wv[(size_t)(d0 + dd) * CCH + t];

    // --- issue async copy of chunk 0 ---
    // each thread moves 4 x 16B: element q = t + 256*i, row = q/64, col16 = q%64
    #pragma unroll
    for (int i = 0; i < 4; ++i) {
        int q   = t + 256 * i;
        int row = q >> 6;
        int col = q & 63;
        unsigned voff = (unsigned)(((0 + row) * NPIX + n0 + col * 4) * 4);
        unsigned ldsa = (unsigned)(uintptr_t)(void*)&xs[0][q * 4];
        asm volatile("global_load_async_to_lds_b128 %0, %1, %2"
                     :: "v"(ldsa), "v"(voff), "s"(xb) : "memory");
    }

    float acc[32];
    for (int dd = 0; dd < 32; ++dd) acc[dd] = bv[d0 + dd];

    for (int ck = 0; ck < NSTAGE; ++ck) {
        const int buf = ck & 1;
        if (ck + 1 < NSTAGE) {                     // issue next chunk
            const int c1 = (ck + 1) * VCHUNK;
            #pragma unroll
            for (int i = 0; i < 4; ++i) {
                int q   = t + 256 * i;
                int row = q >> 6;
                int col = q & 63;
                unsigned voff = (unsigned)(((c1 + row) * NPIX + n0 + col * 4) * 4);
                unsigned ldsa = (unsigned)(uintptr_t)(void*)&xs[buf ^ 1][q * 4];
                asm volatile("global_load_async_to_lds_b128 %0, %1, %2"
                             :: "v"(ldsa), "v"(voff), "s"(xb) : "memory");
            }
            asm volatile("s_wait_asynccnt 0x4" ::: "memory");  // chunk ck done
        } else {
            asm volatile("s_wait_asynccnt 0x0" ::: "memory");
        }
        __syncthreads();                           // all waves' copies visible
        const int c0 = ck * VCHUNK;
        for (int cc = 0; cc < VCHUNK; ++cc) {
            float xv = xs[buf][cc * 256 + t];      // conflict-free, stride 1
            const float* wrow = &wv[(c0 + cc) * 32];  // broadcast
            #pragma unroll
            for (int dd = 0; dd < 32; ++dd) acc[dd] += wrow[dd] * xv;
        }
        __syncthreads();                           // before buf is re-filled
    }

    #pragma unroll
    for (int dd = 0; dd < 32; ++dd)
        vbf[((size_t)b * CCH + d0 + dd) * NPIX + n] = (__bf16)acc[dd];
}

// ---------------------------------------------------------------------------
// Kernel 3: energy[b,n,m] = sum_d q[b,n,d]*k[b,m,d]
// One 16x16 tile per wave; DK=32 => exactly one v_wmma_f32_16x16x32_bf16.
// ---------------------------------------------------------------------------
__global__ __launch_bounds__(256)
void energy_kernel(const __bf16* __restrict__ qbf, const __bf16* __restrict__ kbf,
                   float* __restrict__ energy) {
    const int lane = threadIdx.x & 31;
    const int wave = threadIdx.x >> 5;
    const int l15  = lane & 15;
    const int hi   = lane >> 4;

    const long tile = (long)blockIdx.x * 8 + wave;        // B*256*256 tiles
    const int  b    = (int)(tile / (256L * 256L));
    const int  rem  = (int)(tile % (256L * 256L));
    const int  n0   = (rem >> 8) << 4;
    const int  m0   = (rem & 255) << 4;

    const __bf16* qb = qbf + (size_t)b * NPIX * DKQ;
    const __bf16* kb = kbf + (size_t)b * NPIX * DKQ;

    v8bf alo = *(const v8bf*)(qb + (size_t)(n0 + l15) * DKQ + 16 * hi);
    v8bf ahi = *(const v8bf*)(qb + (size_t)(n0 + l15) * DKQ + 16 * hi + 8);
    v16bf a;
    #pragma unroll
    for (int i = 0; i < 8; ++i) { a[i] = alo[i]; a[8 + i] = ahi[i]; }

    v16bf bm = *(const v16bf*)(kb + (size_t)(m0 + l15) * DKQ + 16 * hi);

    v8f c = {};
    c = __builtin_amdgcn_wmma_f32_16x16x32_bf16(false, a, false, bm,
                                                (short)0, c, false, false);

    float* eb = energy + (size_t)b * NPIX * NPIX;
    #pragma unroll
    for (int r = 0; r < 8; ++r)
        eb[(size_t)(n0 + r + 8 * hi) * NPIX + m0 + l15] = c[r];
}

// ---------------------------------------------------------------------------
// Kernel 4: in-place row softmax over 4096 columns (one block per row).
// ---------------------------------------------------------------------------
__global__ __launch_bounds__(256)
void softmax_kernel(float* __restrict__ attn) {
    __shared__ float red[256];
    const int t = threadIdx.x;
    float* p = attn + (size_t)blockIdx.x * NPIX;

    float vals[16];
    float m = -INFINITY;
    #pragma unroll
    for (int i = 0; i < 16; ++i) { vals[i] = p[t + i * 256]; m = fmaxf(m, vals[i]); }
    red[t] = m;
    __syncthreads();
    for (int s = 128; s > 0; s >>= 1) {
        if (t < s) red[t] = fmaxf(red[t], red[t + s]);
        __syncthreads();
    }
    m = red[0];
    __syncthreads();

    float sum = 0.f;
    #pragma unroll
    for (int i = 0; i < 16; ++i) { vals[i] = __expf(vals[i] - m); sum += vals[i]; }
    red[t] = sum;
    __syncthreads();
    for (int s = 128; s > 0; s >>= 1) {
        if (t < s) red[t] += red[t + s];
        __syncthreads();
    }
    const float inv = 1.0f / red[0];
    #pragma unroll
    for (int i = 0; i < 16; ++i) p[t + i * 256] = vals[i] * inv;
}

// ---------------------------------------------------------------------------
// Kernel 5: out[b,d,n] = gamma * sum_m v[b,d,m]*attn[b,n,m] + x[b,d,n]
// Block-cooperative: 8 waves share one n-tile (16 cols) and cover 128
// channels.  The 16x64 fp32 attn tile is staged + converted to bf16 in LDS
// ONCE per block per K-step (8x fewer cvts / attn bytes than per-wave),
// B-fragments come from LDS, A-fragments (v, bf16) stream from global with
// prefetch.  4096/32 = 128 WMMAs per wave.
// ---------------------------------------------------------------------------
__global__ __launch_bounds__(256)
void pv_kernel(const __bf16* __restrict__ vbf, const float* __restrict__ attn,
               const float* __restrict__ x, const float* __restrict__ gamma,
               float* __restrict__ out) {
    __shared__ __bf16 sb[16][64];                  // 2KB attn tile (n x m)
    const int t    = threadIdx.x;
    const int lane = t & 31;
    const int wave = t >> 5;
    const int l15  = lane & 15;
    const int hi   = lane >> 4;

    const int n0 = blockIdx.x * 16;                // 256 n-tiles
    const int d0 = blockIdx.y * 128 + wave * 16;   // 2 channel blocks x 8 waves
    const int b  = blockIdx.z;

    // staging coordinates: thread t moves 4 floats of row sr
    const int sr = t >> 4;                         // 0..15 (n within tile)
    const int sc = (t & 15) * 4;                   // 0..60 (m within 64-step)
    const float* srow = attn + ((size_t)b * NPIX + n0 + sr) * NPIX + sc;

    const __bf16* vrow = vbf + ((size_t)b * CCH + d0 + l15) * NPIX;

    v8f acc = {};
    for (int km = 0; km < NPIX; km += 64) {
        // ---- cooperative stage: 16x64 attn fp32 -> bf16 in LDS ----
        v4f f = *(const v4f*)(srow + km);
        if (km + 256 < NPIX) __builtin_prefetch(srow + km + 256, 0, 1);
        v4bf h;
        #pragma unroll
        for (int i = 0; i < 4; ++i) h[i] = (__bf16)f[i];
        *(v4bf*)&sb[sr][sc] = h;
        __syncthreads();

        // ---- 2 x K=32 WMMA steps from LDS (B) + global bf16 v (A) ----
        #pragma unroll
        for (int s = 0; s < 2; ++s) {
            const int ks = km + 32 * s;
            v8bf alo = *(const v8bf*)(vrow + ks + 16 * hi);
            v8bf ahi = *(const v8bf*)(vrow + ks + 16 * hi + 8);
            v16bf a;
            #pragma unroll
            for (int i = 0; i < 8; ++i) { a[i] = alo[i]; a[8 + i] = ahi[i]; }
            v16bf bm = *(const v16bf*)&sb[l15][32 * s + 16 * hi];
            acc = __builtin_amdgcn_wmma_f32_16x16x32_bf16(false, a, false, bm,
                                                          (short)0, acc,
                                                          false, false);
        }
        if (km + 256 < NPIX) __builtin_prefetch(vrow + km + 256, 0, 1);
        __syncthreads();                           // sb reuse next step
    }

    const float g = gamma[0];
    #pragma unroll
    for (int r = 0; r < 8; ++r) {
        size_t idx = ((size_t)b * CCH + d0 + r + 8 * hi) * NPIX + n0 + l15;
        out[idx] = g * acc[r] + x[idx];
    }
}

// ---------------------------------------------------------------------------
extern "C" void kernel_launch(void* const* d_in, const int* in_sizes, int n_in,
                              void* d_out, int out_size, void* d_ws, size_t ws_size,
                              hipStream_t stream) {
    const float* x     = (const float*)d_in[0];
    const float* Wq    = (const float*)d_in[1];
    const float* bq    = (const float*)d_in[2];
    const float* Wk    = (const float*)d_in[3];
    const float* bk    = (const float*)d_in[4];
    const float* Wv    = (const float*)d_in[5];
    const float* bv    = (const float*)d_in[6];
    const float* gamma = (const float*)d_in[7];

    float* out  = (float*)d_out;                         // [B,C,N] fp32
    float* attn = out + (size_t)BATCH * CCH * NPIX;      // [B,N,N] fp32

    __bf16* qbf = (__bf16*)d_ws;                         // 1 MB
    __bf16* kbf = qbf + (size_t)BATCH * NPIX * DKQ;      // 1 MB
    __bf16* vbf = kbf + (size_t)BATCH * NPIX * DKQ;      // 8 MB

    proj_qk_kernel<<<BATCH * (NPIX / 256), 256, 0, stream>>>(x, Wq, bq, Wk, bk, qbf, kbf);
    proj_v_kernel<<<dim3(BATCH * (NPIX / 256), CCH / 32), 256, 0, stream>>>(x, Wv, bv, vbf);
    energy_kernel<<<(BATCH * 256 * 256) / 8, 256, 0, stream>>>(qbf, kbf, attn);
    softmax_kernel<<<BATCH * NPIX, 256, 0, stream>>>(attn);
    pv_kernel<<<dim3(NPIX / 16, CCH / 128, BATCH), 256, 0, stream>>>(vbf, attn, x, gamma, out);
}
